// TriangleAttentionCore_13280038879339
// MI455X (gfx1250) — matrix operand order repaired and checked by reading, not compiled, not run on hardware
//
#include <hip/hip_runtime.h>
#include <hip/hip_bf16.h>
#include <math.h>

// ---------------------------------------------------------------------------
// TriangleAttentionCore for MI455X (gfx1250, wave32, WMMA bf16 16x16x32).
// Memory-bound (~370MB traffic vs ~27 GFLOP) -> bf16 intermediates,
// fp32 LN / softmax / accumulation via v_wmma_f32_16x16x32_bf16.
// ---------------------------------------------------------------------------

typedef __bf16 bf16_t;
typedef __attribute__((ext_vector_type(16))) __bf16 v16bf;
typedef __attribute__((ext_vector_type(8)))  __bf16 v8bf;
typedef __attribute__((ext_vector_type(8)))  float  v8f;

#define NSEQ 320
#define CZ   128
#define CH   32
#define NH   4
#define DHID 128   // NH*CH

__device__ __forceinline__ v8f zero8() {
  v8f z;
#pragma unroll
  for (int i = 0; i < 8; ++i) z[i] = 0.0f;
  return z;
}

// A fragment (16x32 bf16, M x K), source row-major [M][K] with given element
// stride.  ISA 7.12.2: lanes 0-15 row M=l16 K0..7 (elems 0-7) / K16..23
// (elems 8-15); lanes 16-31 K8..15 / K24..31.
__device__ __forceinline__ v16bf load_afrag(const bf16_t* base, int stride, int lane) {
  const int l16 = lane & 15, hi = (lane >> 4) & 1;
  const bf16_t* p = base + l16 * stride + hi * 8;
  v8bf lo = *reinterpret_cast<const v8bf*>(p);
  v8bf hh = *reinterpret_cast<const v8bf*>(p + 16);
  return __builtin_shufflevector(lo, hh, 0,1,2,3,4,5,6,7,8,9,10,11,12,13,14,15);
}

// B fragment (32x16 bf16, K x N), sourced from B-transposed storage
// (row-major [N][K], stride in elements).  Lane l16 = column N, elems 0-7 hold
// K = hi*8 + e, elems 8-15 hold K = 16 + hi*8 + (e-8) (same order as A).
__device__ __forceinline__ v16bf load_bfrag(const bf16_t* baseT, int stride, int lane) {
  const int l16 = lane & 15, hi = (lane >> 4) & 1;
  const bf16_t* p = baseT + l16 * stride + hi * 8;
  v8bf lo = *reinterpret_cast<const v8bf*>(p);
  v8bf hh = *reinterpret_cast<const v8bf*>(p + 16);
  return __builtin_shufflevector(lo, hh, 0,1,2,3,4,5,6,7,8,9,10,11,12,13,14,15);
}

__device__ __forceinline__ v8f wmma_bf16(v16bf a, v16bf b, v8f c) {
  return __builtin_amdgcn_wmma_f32_16x16x32_bf16(false, a, false, b, (short)0, c,
                                                 false, false);
}

// Reduce across the 16 lanes holding one matrix row (lane-halves independent).
__device__ __forceinline__ float half_max(float v) {
  v = fmaxf(v, __shfl_xor(v, 1, 32));
  v = fmaxf(v, __shfl_xor(v, 2, 32));
  v = fmaxf(v, __shfl_xor(v, 4, 32));
  v = fmaxf(v, __shfl_xor(v, 8, 32));
  return v;
}
__device__ __forceinline__ float half_sum(float v) {
  v += __shfl_xor(v, 1, 32);
  v += __shfl_xor(v, 2, 32);
  v += __shfl_xor(v, 4, 32);
  v += __shfl_xor(v, 8, 32);
  return v;
}

// ---------------------------------------------------------------------------
// k0: convert the 5 fp32 weight matrices (128x128 each) to bf16 in ws.
// ---------------------------------------------------------------------------
__global__ void k0_cvt_weights(const float* __restrict__ wq, const float* __restrict__ wk,
                               const float* __restrict__ wv, const float* __restrict__ wg,
                               const float* __restrict__ wo, bf16_t* __restrict__ dst) {
  int idx = blockIdx.x * blockDim.x + threadIdx.x;
  if (idx >= 5 * CZ * DHID) return;
  int which = idx >> 14, off = idx & 16383;
  const float* s = (which == 0) ? wq : (which == 1) ? wk : (which == 2) ? wv
                 : (which == 3) ? wg : wo;
  dst[idx] = (bf16_t)s[off];
}

// ---------------------------------------------------------------------------
// k1: LayerNorm(z) -> zn (bf16, LDS), then fused q/k/v/g projections.
// Grid (20, 320): 16-row j-tile per block, 128 threads = 4 waves; wave w owns
// weight matrix w (q,k,v,g).  q pre-scaled by 1/sqrt(32); g pre-sigmoided;
// v stored transposed [i][h][c][j] via an LDS staging tile.
// ---------------------------------------------------------------------------
__global__ __launch_bounds__(128) void k1_ln_proj(
    const float* __restrict__ z, const float* __restrict__ lnw,
    const float* __restrict__ lnb, const bf16_t* __restrict__ wb,
    bf16_t* __restrict__ q_ws, bf16_t* __restrict__ k_ws,
    bf16_t* __restrict__ vT_ws, bf16_t* __restrict__ g_ws) {
  const int jt = blockIdx.x, i = blockIdx.y;
  const int j0 = jt * 16;
  const int tid = threadIdx.x;

  __shared__ float s_sum[16][8];
  __shared__ float s_sq[16][8];
  __shared__ float s_mu[16];
  __shared__ float s_rs[16];
  __shared__ alignas(32) bf16_t zn[16 * CZ];      // 4 KB
  __shared__ float vstage[16][17];                // v-transpose staging (wave 2)

  // ---- LayerNorm: 8 threads x 16 channels per row ----
  {
    const int r = tid >> 3, c0 = (tid & 7) * 16;
    const float* zp = z + ((size_t)(i * NSEQ + j0 + r)) * CZ + c0;
    float x[16];
    float s = 0.0f, q = 0.0f;
#pragma unroll
    for (int u = 0; u < 16; ++u) { x[u] = zp[u]; s += x[u]; q += x[u] * x[u]; }
    s_sum[r][tid & 7] = s;
    s_sq[r][tid & 7] = q;
    __syncthreads();
    if (tid < 16) {
      float S = 0.0f, Q = 0.0f;
#pragma unroll
      for (int u = 0; u < 8; ++u) { S += s_sum[tid][u]; Q += s_sq[tid][u]; }
      float mu = S * (1.0f / CZ);
      float var = Q * (1.0f / CZ) - mu * mu;
      s_mu[tid] = mu;
      s_rs[tid] = rsqrtf(var + 1e-5f);
    }
    __syncthreads();
    const float mu = s_mu[r], rs = s_rs[r];
#pragma unroll
    for (int u = 0; u < 16; ++u) {
      float v = (x[u] - mu) * rs * lnw[c0 + u] + lnb[c0 + u];
      zn[r * CZ + c0 + u] = (bf16_t)v;
    }
  }
  __syncthreads();

  // ---- projections: wave w -> weight w; 8 N-tiles x 4 K-steps of WMMA ----
  const int wave = tid >> 5, lane = tid & 31;
  const int l16 = lane & 15, hi = (lane >> 4) & 1;

  v16bf aZ[4];
#pragma unroll
  for (int kt = 0; kt < 4; ++kt)
    aZ[kt] = load_afrag(zn + kt * 32, CZ, lane);

  const bf16_t* Wt = wb + wave * (CZ * DHID);   // row-major [e][d] == B^T
  const float qscale = 0.17677669529663689f;    // 1/sqrt(32)

  for (int nt = 0; nt < 8; ++nt) {
    v8f acc = zero8();
#pragma unroll
    for (int kt = 0; kt < 4; ++kt) {
      v16bf b = load_bfrag(Wt + nt * 16 * CZ + kt * 32, CZ, lane);
      acc = wmma_bf16(aZ[kt], b, acc);
    }
    const int h = nt >> 1, csub = (nt & 1) * 16;
    if (wave == 0 || wave == 1) {               // q (scaled) / k
      bf16_t* dst = (wave == 0) ? q_ws : k_ws;
      const float sc = (wave == 0) ? qscale : 1.0f;
#pragma unroll
      for (int r = 0; r < 8; ++r) {
        int j = j0 + r + hi * 8;
        dst[((size_t)(i * NH + h) * NSEQ + j) * CH + csub + l16] = (bf16_t)(acc[r] * sc);
      }
    } else if (wave == 3) {                     // g = sigmoid(zn @ wg)
#pragma unroll
      for (int r = 0; r < 8; ++r) {
        int j = j0 + r + hi * 8;
        float gv = 1.0f / (1.0f + __expf(-acc[r]));
        g_ws[((size_t)(i * NSEQ + j)) * DHID + nt * 16 + l16] = (bf16_t)gv;
      }
    } else {                                    // v -> transposed [c][j]
#pragma unroll
      for (int r = 0; r < 8; ++r)
        vstage[l16][r + hi * 8] = acc[r];       // wave-local; hw waits ds
      const int c2 = lane >> 1, part = lane & 1;
      bf16_t* dst = vT_ws + ((size_t)(i * NH + h) * CH + csub + c2) * NSEQ +
                    j0 + part * 8;
      v8bf tmp;
#pragma unroll
      for (int u = 0; u < 8; ++u) tmp[u] = (bf16_t)vstage[c2][part * 8 + u];
      *reinterpret_cast<v8bf*>(dst) = tmp;
    }
  }
}

// ---------------------------------------------------------------------------
// k2: attention per (i,h).  Grid (4, 320), 128 threads = 4 waves; each wave
// owns 5 q-tiles of 16 rows.  Scores: 1 WMMA per 16x16 tile (K=c=32).
// Softmax in fp32 (unnormalized-PV trick), P tile staged in 16x32 bf16 LDS,
// PV: 10 WMMAs per output column-tile.  Gating fused at store.
// ---------------------------------------------------------------------------
__global__ __launch_bounds__(128) void k2_attn(
    const float* __restrict__ mask, const float* __restrict__ edge,
    const bf16_t* __restrict__ q_ws, const bf16_t* __restrict__ k_ws,
    const bf16_t* __restrict__ vT_ws, const bf16_t* __restrict__ g_ws,
    bf16_t* __restrict__ o_ws) {
  const int h = blockIdx.x, i = blockIdx.y;
  const int tid = threadIdx.x;
  const int wave = tid >> 5, lane = tid & 31;
  const int l16 = lane & 15, hi = (lane >> 4) & 1;

  __shared__ alignas(64) bf16_t ks[NSEQ * CH];        // 20 KB  [j][c]
  __shared__ alignas(64) bf16_t vTs[CH * NSEQ];       // 20 KB  [c][j]
  __shared__ alignas(64) bf16_t ptile[4][16 * 32];    //  4 KB  per-wave P tile

  // cooperative 16B copies into LDS
  {
    const uint4* ksrc = reinterpret_cast<const uint4*>(
        k_ws + (size_t)(i * NH + h) * NSEQ * CH);
    const uint4* vsrc = reinterpret_cast<const uint4*>(
        vT_ws + (size_t)(i * NH + h) * CH * NSEQ);
    uint4* kdst = reinterpret_cast<uint4*>(ks);
    uint4* vdst = reinterpret_cast<uint4*>(vTs);
#pragma unroll
    for (int u = 0; u < 10; ++u) {
      kdst[tid + u * 128] = ksrc[tid + u * 128];
      vdst[tid + u * 128] = vsrc[tid + u * 128];
    }
  }
  __syncthreads();

  for (int it = 0; it < 5; ++it) {
    const int qt = wave * 5 + it;
    const int j0 = qt * 16;

    // Q fragment straight from global (reused across all 20 k-tiles)
    v16bf aQ = load_afrag(q_ws + ((size_t)(i * NH + h) * NSEQ + j0) * CH, CH, lane);

    // scores
    v8f sc[20];
#pragma unroll
    for (int kt = 0; kt < 20; ++kt) {
      v16bf bk = load_bfrag(ks + kt * 16 * CH, CH, lane);
      sc[kt] = wmma_bf16(aQ, bk, zero8());
    }

    // + mask_bias[i, k] + edge_bias[h, q, k]
#pragma unroll
    for (int kt = 0; kt < 20; ++kt) {
      const int kcol = kt * 16 + l16;
      const float mb = mask[i * NSEQ + kcol];
#pragma unroll
      for (int r = 0; r < 8; ++r) {
        const int q = j0 + r + hi * 8;
        sc[kt][r] += mb + edge[((size_t)(h * NSEQ) + q) * NSEQ + kcol];
      }
    }

    // row max (16-lane half-reduction)
    float mx[8];
#pragma unroll
    for (int r = 0; r < 8; ++r) mx[r] = -3.0e38f;
#pragma unroll
    for (int kt = 0; kt < 20; ++kt)
#pragma unroll
      for (int r = 0; r < 8; ++r) mx[r] = fmaxf(mx[r], sc[kt][r]);
#pragma unroll
    for (int r = 0; r < 8; ++r) mx[r] = half_max(mx[r]);

    // exp -> LDS P tile (K=32 chunk) -> PV WMMA, accumulate unnormalized
    float sm[8];
#pragma unroll
    for (int r = 0; r < 8; ++r) sm[r] = 0.0f;
    v8f oacc[2] = {zero8(), zero8()};

    for (int kt2 = 0; kt2 < 10; ++kt2) {
#pragma unroll
      for (int u = 0; u < 2; ++u) {
        const int kt = 2 * kt2 + u;
#pragma unroll
        for (int r = 0; r < 8; ++r) {
          float p = __expf(sc[kt][r] - mx[r]);
          sm[r] += p;
          ptile[wave][(r + hi * 8) * 32 + u * 16 + l16] = (bf16_t)p;
        }
      }
      v16bf aP = load_afrag(ptile[wave], 32, lane);
#pragma unroll
      for (int ct = 0; ct < 2; ++ct) {
        v16bf bv = load_bfrag(vTs + (ct * 16) * NSEQ + kt2 * 32, NSEQ, lane);
        oacc[ct] = wmma_bf16(aP, bv, oacc[ct]);
      }
    }

    float inv[8];
#pragma unroll
    for (int r = 0; r < 8; ++r) inv[r] = 1.0f / half_sum(sm[r]);

    // normalize, gate, store o (bf16 [i*N+j][128])
#pragma unroll
    for (int ct = 0; ct < 2; ++ct) {
#pragma unroll
      for (int r = 0; r < 8; ++r) {
        const int j = j0 + r + hi * 8;
        const int d = h * CH + ct * 16 + l16;
        const size_t idx = ((size_t)(i * NSEQ + j)) * DHID + d;
        const float gv = (float)g_ws[idx];
        o_ws[idx] = (bf16_t)(oacc[ct][r] * inv[r] * gv);
      }
    }
  }
}

// ---------------------------------------------------------------------------
// k3: out = o_gated @ wo^T.  Rows 102400, K=128 (4 WMMA steps), N=128.
// Grid 6400 x 128 threads; wave w owns N-tiles {2w, 2w+1}.  fp32 output.
// ---------------------------------------------------------------------------
__global__ __launch_bounds__(128) void k3_outproj(
    const bf16_t* __restrict__ o_ws, const bf16_t* __restrict__ wo_bf,
    float* __restrict__ out) {
  const int row0 = blockIdx.x * 16;
  const int tid = threadIdx.x;
  const int wave = tid >> 5, lane = tid & 31;
  const int l16 = lane & 15, hi = (lane >> 4) & 1;

  v16bf aO[4];
#pragma unroll
  for (int kt = 0; kt < 4; ++kt)
    aO[kt] = load_afrag(o_ws + (size_t)row0 * DHID + kt * 32, DHID, lane);

#pragma unroll
  for (int s = 0; s < 2; ++s) {
    const int nt = 2 * wave + s;
    v8f acc = zero8();
#pragma unroll
    for (int kt = 0; kt < 4; ++kt) {
      v16bf b = load_bfrag(wo_bf + nt * 16 * DHID + kt * 32, DHID, lane);
      acc = wmma_bf16(aO[kt], b, acc);
    }
#pragma unroll
    for (int r = 0; r < 8; ++r)
      out[(size_t)(row0 + r + hi * 8) * CZ + nt * 16 + l16] = acc[r];
  }
}

// ---------------------------------------------------------------------------
extern "C" void kernel_launch(void* const* d_in, const int* in_sizes, int n_in,
                              void* d_out, int out_size, void* d_ws, size_t ws_size,
                              hipStream_t stream) {
  (void)in_sizes; (void)n_in; (void)out_size; (void)ws_size;
  const float* z    = (const float*)d_in[0];
  const float* mask = (const float*)d_in[1];
  const float* edge = (const float*)d_in[2];
  const float* wq   = (const float*)d_in[3];
  const float* wk   = (const float*)d_in[4];
  const float* wv   = (const float*)d_in[5];
  const float* wg   = (const float*)d_in[6];
  const float* wo   = (const float*)d_in[7];
  const float* lnw  = (const float*)d_in[8];
  const float* lnb  = (const float*)d_in[9];
  float* out = (float*)d_out;

  // workspace layout (bytes); all 256-aligned
  constexpr size_t W_OFF  = 0;                               // 5x128x128 bf16
  constexpr size_t W_BYT  = (size_t)5 * CZ * DHID * 2;       // 163840
  constexpr size_t QKV_B  = (size_t)NSEQ * NH * NSEQ * CH * 2;  // 26,214,400
  constexpr size_t GO_B   = (size_t)NSEQ * NSEQ * DHID * 2;     // 26,214,400
  char* ws = (char*)d_ws;
  bf16_t* wb    = (bf16_t*)(ws + W_OFF);
  bf16_t* q_ws  = (bf16_t*)(ws + W_BYT);
  bf16_t* k_ws  = (bf16_t*)(ws + W_BYT + QKV_B);
  bf16_t* vT_ws = (bf16_t*)(ws + W_BYT + 2 * QKV_B);
  bf16_t* g_ws  = (bf16_t*)(ws + W_BYT + 3 * QKV_B);
  bf16_t* o_ws  = (bf16_t*)(ws + W_BYT + 3 * QKV_B + GO_B);
  bf16_t* wo_bf = wb + 4 * CZ * DHID;

  k0_cvt_weights<<<(5 * CZ * DHID + 255) / 256, 256, 0, stream>>>(
      wq, wk, wv, wg, wo, wb);
  k1_ln_proj<<<dim3(NSEQ / 16, NSEQ), 128, 0, stream>>>(
      z, lnw, lnb, wb, q_ws, k_ws, vT_ws, g_ws);
  k2_attn<<<dim3(NH, NSEQ), 128, 0, stream>>>(
      mask, edge, q_ws, k_ws, vT_ws, g_ws, o_ws);
  k3_outproj<<<(NSEQ * NSEQ) / 16, 128, 0, stream>>>(o_ws, wo_bf, out);
}
